// ConfounderEncoder_71244917506476
// MI455X (gfx1250) — compile-verified, matrix operand order checked by reading, not run on hardware
//
#include <hip/hip_runtime.h>
#include <math.h>

// Problem constants (match reference)
#define B_  256
#define T_  128
#define V_  8192
#define L_  14
#define LL_ (L_ * L_)   // 196
#define ZD_ 512
#define ZL_ 256
#define ZC_ 256
#define H_  512

typedef __attribute__((ext_vector_type(2)))  float  v2f;
typedef __attribute__((ext_vector_type(8)))  float  v8f;
typedef __attribute__((ext_vector_type(16))) __bf16 v16bf;

// ---------------------------------------------------------------------------
// Kernel 1: bow + pmi_vec.  bow@PMI with <=128 nonzeros/row == gather-sum of
// PMI rows.  One block per batch row; thread tid covers columns tid*4 + j*1024
// so each of the 8 float4 loads per token row is a fully coalesced 4KB burst.
// ---------------------------------------------------------------------------
__global__ __launch_bounds__(256) void bow_pmi_kernel(const int* __restrict__ ids,
                                                      const float* __restrict__ PMI,
                                                      float* __restrict__ pmi_vec) {
    __shared__ int s_ids[T_];
    __shared__ int s_count;
    const int b   = blockIdx.x;
    const int tid = threadIdx.x;
    if (tid == 0) s_count = 0;
    __syncthreads();
    if (tid < T_) {
        int id = ids[b * T_ + tid];
        bool valid = (id >= 0) && (id < V_);
        s_ids[tid] = valid ? id : -1;
        if (valid) atomicAdd(&s_count, 1);
    }
    __syncthreads();

    float4 acc[8];
#pragma unroll
    for (int j = 0; j < 8; ++j) acc[j] = make_float4(0.f, 0.f, 0.f, 0.f);

    for (int t = 0; t < T_; ++t) {
        int id = s_ids[t];
        if (id < 0) continue;
        const float* row = PMI + (size_t)id * V_;
        if (t + 1 < T_) {
            int idn = s_ids[t + 1];
            if (idn >= 0) __builtin_prefetch(PMI + (size_t)idn * V_ + tid * 4, 0, 0);
        }
#pragma unroll
        for (int j = 0; j < 8; ++j) {
            float4 v = *(const float4*)(row + j * 1024 + tid * 4);
            acc[j].x += v.x; acc[j].y += v.y; acc[j].z += v.z; acc[j].w += v.w;
        }
    }

    const float inv = 1.0f / fmaxf((float)s_count, 1.0f);
    float* dst = pmi_vec + (size_t)b * V_;
#pragma unroll
    for (int j = 0; j < 8; ++j) {
        float4 v = make_float4(acc[j].x * inv, acc[j].y * inv, acc[j].z * inv, acc[j].w * inv);
        *(float4*)(dst + j * 1024 + tid * 4) = v;
    }
}

// ---------------------------------------------------------------------------
// Kernel 2: z_lang = pmi_vec @ W_lang^T via WMMA.  One wave per 16x16 output
// tile (16x16 tiles = 256 waves).  Preferred path: exact-precision
// V_WMMA_F32_16X16X4_F32 (A 16x4, B 4x16, fp32 accumulate).
// Fragment layouts per CDNA5 ISA 7.12.2:
//   A(16x4 f32):  lane = m + 16*(k/2), vgpr = k%2
//   B(4x16 f32):  lane = n + 16*(k/2), vgpr = k%2   (N takes the lane role)
//   C(16x16 f32): vgpr r, lane l -> m = r + 8*(l>=16), n = l%16
// ---------------------------------------------------------------------------
__global__ __launch_bounds__(128) void zlang_wmma_kernel(const float* __restrict__ pv,
                                                         const float* __restrict__ wl,
                                                         float* __restrict__ out) {
    const int wave = blockIdx.x * 4 + (threadIdx.x >> 5);
    const int lane = threadIdx.x & 31;
    const int tm   = wave >> 4;          // output row tile   (0..15)
    const int tn   = wave & 15;          // output col tile   (0..15)
    const int half = lane >> 4;          // 0: lanes 0-15, 1: lanes 16-31
    const int lid  = lane & 15;
    const int mrow = tm * 16 + lid;      // A row this lane owns
    const int ncol = tn * 16 + lid;      // B col this lane owns

    v8f acc = {};
    const float* aptr = pv + (size_t)mrow * V_;
    const float* bptr = wl + (size_t)ncol * V_;

#if __has_builtin(__builtin_amdgcn_wmma_f32_16x16x4_f32)
    const float* a2 = aptr + half * 2;
    const float* b2 = bptr + half * 2;
    for (int k0 = 0; k0 < V_; k0 += 4) {
        float2 av = *(const float2*)(a2 + k0);
        float2 bv = *(const float2*)(b2 + k0);
        v2f a; a.x = av.x; a.y = av.y;
        v2f b; b.x = bv.x; b.y = bv.y;
        acc = __builtin_amdgcn_wmma_f32_16x16x4_f32(false, a, false, b,
                                                    (short)0, acc, false, false);
    }
#else
    // Fallback: bf16 16x16x32 WMMA (codegen-confirmed builtin), on-the-fly
    // f32->bf16 conversion.  16-bit A layout: lane half 0 holds K=0..7,16..23;
    // half 1 holds K=8..15,24..31 (element e of v16bf -> vgpr e/2, hi/lo e%2).
    for (int k0 = 0; k0 < V_; k0 += 32) {
        v16bf a, b;
#pragma unroll
        for (int e = 0; e < 8; ++e) {
            a[e]     = (__bf16)aptr[k0 + half * 8 + e];
            a[8 + e] = (__bf16)aptr[k0 + 16 + half * 8 + e];
            b[e]     = (__bf16)bptr[k0 + half * 8 + e];
            b[8 + e] = (__bf16)bptr[k0 + 16 + half * 8 + e];
        }
        acc = __builtin_amdgcn_wmma_f32_16x16x32_bf16(false, a, false, b,
                                                      (short)0, acc, false, false);
    }
#endif

    const int orow = tm * 16 + half * 8;
    const int ocol = tn * 16 + lid;
#pragma unroll
    for (int r = 0; r < 8; ++r) {
        out[(size_t)(orow + r) * ZD_ + ocol] = acc[r];
    }
}

// ---------------------------------------------------------------------------
// Kernel 3: corr MLP (exact GELU) + broadcast into out[:, 256:512].
// One block, 512 threads.
// ---------------------------------------------------------------------------
__global__ __launch_bounds__(512) void corr_mlp_kernel(const float* __restrict__ corr,
                                                       const float* __restrict__ W1,
                                                       const float* __restrict__ b1,
                                                       const float* __restrict__ W2,
                                                       const float* __restrict__ b2,
                                                       float* __restrict__ out) {
    __shared__ float s_corr[LL_];
    __shared__ float s_h[H_];
    __shared__ float s_z[ZC_];
    const int tid = threadIdx.x;
    if (tid < LL_) s_corr[tid] = corr[tid];
    __syncthreads();

    {   // h = gelu(corr_flat @ W1^T + b1), exact gelu
        float x = b1[tid];
        const float* w = W1 + (size_t)tid * LL_;
        for (int k = 0; k < LL_; ++k) x += s_corr[k] * w[k];
        s_h[tid] = 0.5f * x * (1.0f + erff(x * 0.70710678118654752f));
    }
    __syncthreads();

    if (tid < ZC_) {
        float z = b2[tid];
        const float* w = W2 + (size_t)tid * H_;
        for (int j = 0; j < H_; ++j) z += s_h[j] * w[j];
        s_z[tid] = z;
    }
    __syncthreads();

    for (int idx = tid; idx < B_ * ZC_; idx += 512) {
        int b = idx >> 8;       // /256
        int i = idx & 255;
        out[(size_t)b * ZD_ + ZL_ + i] = s_z[i];
    }
}

// ---------------------------------------------------------------------------
extern "C" void kernel_launch(void* const* d_in, const int* in_sizes, int n_in,
                              void* d_out, int out_size, void* d_ws, size_t ws_size,
                              hipStream_t stream) {
    const int*   ids    = (const int*)d_in[0];    // [B,T]
    const float* PMI    = (const float*)d_in[1];  // [V,V]
    const float* W_lang = (const float*)d_in[2];  // [ZL,V]
    const float* corr   = (const float*)d_in[3];  // [L,L]
    const float* W1     = (const float*)d_in[4];  // [H, L*L]
    const float* b1     = (const float*)d_in[5];  // [H]
    const float* W2     = (const float*)d_in[6];  // [ZC,H]
    const float* b2     = (const float*)d_in[7];  // [ZC]
    float* out     = (float*)d_out;               // [B, ZD]
    float* pmi_vec = (float*)d_ws;                // [B, V] scratch (8 MB)

    bow_pmi_kernel<<<B_, 256, 0, stream>>>(ids, PMI, pmi_vec);
    zlang_wmma_kernel<<<64, 128, 0, stream>>>(pmi_vec, W_lang, out);
    corr_mlp_kernel<<<1, 512, 0, stream>>>(corr, W1, b1, W2, b2, out);
}
